// PraxisScatter_58334245814929
// MI455X (gfx1250) — compile-verified, matrix operand order checked by reading, not run on hardware
//
#include <hip/hip_runtime.h>
#include <hip/hip_bf16.h>

// ---------------------------------------------------------------------------
// Problem constants (from reference): B=8, S=1024, D=1024, H=4096
// ---------------------------------------------------------------------------
#define PB 8
#define PS 1024
#define PD 1024
#define PH 4096
#define PM (PB * PS)                 // 8192 rows
#define PER_BATCH ((size_t)PS * PH)  // 4,194,304 scores per batch
#define TOPK_COUNT 1048576u          // min(1024*1024, S*H)

// Set to 0 if the async-to-LDS mnemonic is rejected by the assembler.
#define USE_ASYNC_LDS 1

typedef __attribute__((ext_vector_type(16))) __bf16 v16bf;
typedef __attribute__((ext_vector_type(8)))  __bf16 v8bf;
typedef __attribute__((ext_vector_type(8)))  float  v8f;

enum { MODE_RELU_BF16 = 0, MODE_SCORES_F32 = 1, MODE_SELECT_SILU = 2, MODE_PLAIN_F32 = 3 };

// Monotone key: larger float -> larger unsigned
__device__ __forceinline__ unsigned mono_key(float f) {
  unsigned u = __float_as_uint(f);
  return u ^ ((u & 0x80000000u) ? 0xFFFFFFFFu : 0x80000000u);
}

// ---------------------------------------------------------------------------
// Async global->LDS 16-byte copy (ASYNCcnt-tracked), per-lane addresses.
// ---------------------------------------------------------------------------
__device__ __forceinline__ void async_copy16(unsigned lds_off, const void* g) {
#if USE_ASYNC_LDS
  asm volatile("global_load_async_to_lds_b128 %0, %1, off"
               :: "v"(lds_off), "v"((unsigned long long)(uintptr_t)g)
               : "memory");
#else
  // sync fallback: regular load + ds store (compiler addresses LDS via offset)
  *(v8bf*)((__attribute__((address_space(3))) char*)0 + lds_off) = *(const v8bf*)g; // unused
#endif
}

__device__ __forceinline__ void wait_async0() {
#if USE_ASYNC_LDS
  asm volatile("s_wait_asynccnt 0x0" ::: "memory");
#endif
}

// Build a v16bf WMMA fragment from two contiguous 16B chunks (ISA 16-bit
// 16x32 layout: lanes 0-15 take K=koff..koff+7 and koff+16..koff+23).
__device__ __forceinline__ v16bf frag_from(const __bf16* p) {
  v8bf lo = *(const v8bf*)(p);
  v8bf hi = *(const v8bf*)(p + 16);
  v16bf r;
#pragma unroll
  for (int e = 0; e < 8; ++e) { r[e] = lo[e]; r[e + 8] = hi[e]; }
  return r;
}

// ---------------------------------------------------------------------------
// Workgroup GEMM: 256 threads = 8 waves (2x4), macro tile 128x128.
// Each wave owns a 64x32 sub-tile = 4x2 WMMA 16x16 tiles.
// A[M,K] * W[N,K]^T + bias, bf16 inputs, f32 accumulate, K-step 32.
// Double-buffered LDS staging via async global->LDS copies.
// LDS rows padded to 40 bf16 (80B) -> conflict-free ds_load_b128 frag reads.
// ---------------------------------------------------------------------------
#define LROW 40
#define BUF_BYTES (2 * 128 * LROW * 2)  // A-tile + W-tile, one buffer

template <int MODE>
__global__ __launch_bounds__(256) void gemm128_bf16(
    const __bf16* __restrict__ A,
    const __bf16* __restrict__ W0,
    const __bf16* __restrict__ W1,
    const float*  __restrict__ bias0,
    const float*  __restrict__ bias1,
    const int*    __restrict__ mask,   // [B,H] (SELECT mode)
    void* __restrict__ out,
    int M, int N, int K)
{
  __shared__ alignas(16) __bf16 smem[2][2][128][LROW];

  const int t    = threadIdx.x;
  const int lane = t & 31;
  const int wid  = t >> 5;
  const int l15  = lane & 15;
  const int kh   = lane >> 4;
  const int koff = kh * 8;
  const int wr   = wid >> 2;          // 0..1  (row of wave grid)
  const int wc   = wid & 3;           // 0..3  (col of wave grid)
  const int gm0  = blockIdx.y * 128;
  const int gn0  = blockIdx.x * 128;
  const int bidx = gm0 >> 10;         // batch id; 128-row tiles never cross S=1024

  const unsigned sbase = (unsigned)(uintptr_t)&smem[0][0][0][0];

  // Staging assignment: thread t copies 16B chunk cx (0..3) of rows
  // rowA and rowA+64, for both the A tile and the W tile.
  const int rowA = t >> 2;            // 0..63
  const int cx   = t & 3;             // chunk within 32-element row

  const __bf16* aRow[2];
  const __bf16* wRow[2];
  unsigned ldsA[2], ldsW[2];
#pragma unroll
  for (int q = 0; q < 2; ++q) {
    int row = rowA + q * 64;
    aRow[q] = A + (size_t)(gm0 + row) * K + cx * 8;
    const __bf16* wb = W0;
    if (MODE == MODE_SELECT_SILU && mask[bidx * N + gn0 + row]) wb = W1;
    wRow[q] = wb + (size_t)(gn0 + row) * K + cx * 8;
    ldsA[q] = sbase + (unsigned)(((0 * 128 + row) * LROW + cx * 8) * 2);
    ldsW[q] = sbase + (unsigned)(((1 * 128 + row) * LROW + cx * 8) * 2);
  }

  v8f acc[4][2];
#pragma unroll
  for (int i = 0; i < 4; ++i)
#pragma unroll
    for (int j = 0; j < 2; ++j) acc[i][j] = (v8f){};

  const int KT = K >> 5;  // K / 32

  // stage buffer 0
#pragma unroll
  for (int q = 0; q < 2; ++q) {
    async_copy16(ldsA[q], aRow[q]);
    async_copy16(ldsW[q], wRow[q]);
  }

  for (int kt = 0; kt < KT; ++kt) {
    wait_async0();        // my 4 outstanding stage copies for buffer kt&1 done
    __syncthreads();      // all waves: buffer ready; prior-step LDS reads done

    if (kt + 1 < KT) {
      const unsigned boff = (unsigned)(((kt + 1) & 1) * BUF_BYTES);
      const int kc = (kt + 1) << 5;
#pragma unroll
      for (int q = 0; q < 2; ++q) {
        async_copy16(ldsA[q] + boff, aRow[q] + kc);
        async_copy16(ldsW[q] + boff, wRow[q] + kc);
      }
    }

    const int buf = kt & 1;
    v16bf af[4], wf[2];
#pragma unroll
    for (int i = 0; i < 4; ++i)
      af[i] = frag_from(&smem[buf][0][wr * 64 + i * 16 + l15][koff]);
#pragma unroll
    for (int j = 0; j < 2; ++j)
      wf[j] = frag_from(&smem[buf][1][wc * 32 + j * 16 + l15][koff]);

#pragma unroll
    for (int i = 0; i < 4; ++i)
#pragma unroll
      for (int j = 0; j < 2; ++j)
        acc[i][j] = __builtin_amdgcn_wmma_f32_16x16x32_bf16(
            false, af[i], false, wf[j], (short)0, acc[i][j], false, false);
  }

  // Epilogue. C/D layout: VGPR r -> row base + 8*kh + r, col base + l15.
#pragma unroll
  for (int i = 0; i < 4; ++i) {
#pragma unroll
    for (int j = 0; j < 2; ++j) {
      int n = gn0 + wc * 32 + j * 16 + l15;
      float bn;
      if (MODE == MODE_SELECT_SILU)
        bn = mask[bidx * N + n] ? bias1[n] : bias0[n];
      else
        bn = bias0[n];
#pragma unroll
      for (int r = 0; r < 8; ++r) {
        int m = gm0 + wr * 64 + i * 16 + kh * 8 + r;
        float v = acc[i][j][r] + bn;
        size_t idx = (size_t)m * N + n;
        if (MODE == MODE_RELU_BF16) {
          v = fmaxf(v, 0.0f);
          ((__bf16*)out)[idx] = (__bf16)v;
        } else if (MODE == MODE_SELECT_SILU) {
          v = v / (1.0f + __expf(-v));          // silu
          ((__bf16*)out)[idx] = (__bf16)v;
        } else {                                // SCORES_F32 / PLAIN_F32
          ((float*)out)[idx] = v;
        }
      }
    }
  }
}

// ---------------------------------------------------------------------------
// f32 -> bf16 conversion
// ---------------------------------------------------------------------------
__global__ void f2bf_kernel(const float* __restrict__ in, __bf16* __restrict__ out, size_t n) {
  size_t i = (size_t)blockIdx.x * blockDim.x + threadIdx.x;
  if (i < n) out[i] = (__bf16)in[i];
}

// ---------------------------------------------------------------------------
// Radix-select level 0: histogram of top-12 key bits, LDS-privatized.
// ---------------------------------------------------------------------------
__global__ void hist_top12(const float* __restrict__ scores, unsigned* __restrict__ hist) {
  __shared__ unsigned lh[4096];
  const int b = blockIdx.x;
  const int per = (int)(PER_BATCH / gridDim.y);
  const size_t base = (size_t)b * PER_BATCH + (size_t)blockIdx.y * per;
  for (int i = threadIdx.x; i < 4096; i += blockDim.x) lh[i] = 0;
  __syncthreads();
  for (int i = threadIdx.x; i < per; i += blockDim.x) {
    unsigned key = mono_key(scores[base + i]);
    atomicAdd(&lh[key >> 20], 1u);
  }
  __syncthreads();
  for (int i = threadIdx.x; i < 4096; i += blockDim.x) {
    unsigned c = lh[i];
    if (c) atomicAdd(&hist[b * 4096 + i], c);
  }
}

__global__ void refine_hist(const float* __restrict__ scores,
                            const unsigned* __restrict__ state,
                            unsigned* __restrict__ hist,
                            int shift_match, int shift_bin, unsigned binmask) {
  const int b = blockIdx.x;
  const unsigned pref = state[b * 2 + 1];
  const int per = (int)(PER_BATCH / gridDim.y);
  const size_t base = (size_t)b * PER_BATCH + (size_t)blockIdx.y * per;
  for (int i = threadIdx.x; i < per; i += blockDim.x) {
    unsigned key = mono_key(scores[base + i]);
    if ((key >> shift_match) == pref)
      atomicAdd(&hist[b * (binmask + 1) + ((key >> shift_bin) & binmask)], 1u);
  }
}

__global__ void scan_hist(const unsigned* __restrict__ hist, unsigned* __restrict__ state,
                          int nbins, int bits) {
  const int b = blockIdx.x;
  if (threadIdx.x != 0) return;
  unsigned r = state[b * 2], cum = 0;
  int sel = 0;
  for (int bin = nbins - 1; bin >= 0; --bin) {
    unsigned c = hist[b * nbins + bin];
    if (cum + c >= r) { sel = bin; break; }
    cum += c;
  }
  state[b * 2]     = r - cum;
  state[b * 2 + 1] = (state[b * 2 + 1] << bits) | (unsigned)sel;
}

__global__ void init_state(unsigned* __restrict__ state) {
  int b = threadIdx.x;
  if (b < PB) { state[b * 2] = TOPK_COUNT; state[b * 2 + 1] = 0u; }
}

__global__ void colmax_mask(const float* __restrict__ scores,
                            const unsigned* __restrict__ state,
                            int* __restrict__ mask) {
  const int b = blockIdx.y;
  const int h = blockIdx.x * blockDim.x + threadIdx.x;
  const unsigned tkey = state[b * 2 + 1];
  const float* p = scores + (size_t)b * PER_BATCH + h;
  float mx = -3.402823466e38f;
  for (int s = 0; s < PS; ++s) mx = fmaxf(mx, p[(size_t)s * PH]);
  mask[b * PH + h] = (mono_key(mx) >= tkey) ? 1 : 0;
}

// ---------------------------------------------------------------------------
// Host launcher
// ---------------------------------------------------------------------------
extern "C" void kernel_launch(void* const* d_in, const int* in_sizes, int n_in,
                              void* d_out, int out_size, void* d_ws, size_t ws_size,
                              hipStream_t stream) {
  (void)in_sizes; (void)n_in; (void)out_size; (void)ws_size;

  const float* inputs  = (const float*)d_in[0];
  const float* up_w    = (const float*)d_in[1];
  const float* up_b    = (const float*)d_in[2];
  const float* gate_w1 = (const float*)d_in[3];
  const float* gate_b1 = (const float*)d_in[4];
  const float* gate_w2 = (const float*)d_in[5];
  const float* gate_b2 = (const float*)d_in[6];
  const float* mod_w   = (const float*)d_in[7];
  const float* mod_b   = (const float*)d_in[8];
  const float* down_w  = (const float*)d_in[9];
  const float* down_b  = (const float*)d_in[10];
  float* out = (float*)d_out;

  char* wsp = (char*)d_ws;
  auto alloc = [&](size_t bytes) -> void* {
    void* p = (void*)wsp;
    wsp += (bytes + 255) & ~(size_t)255;
    return p;
  };

  __bf16* Xbf  = (__bf16*)alloc((size_t)PM * PD * 2);
  __bf16* g1bf = (__bf16*)alloc((size_t)PH * PD * 2);
  __bf16* g2bf = (__bf16*)alloc((size_t)PH * PH * 2);
  __bf16* upbf = (__bf16*)alloc((size_t)PH * PD * 2);
  __bf16* mdbf = (__bf16*)alloc((size_t)PH * PD * 2);
  __bf16* dnbf = (__bf16*)alloc((size_t)PD * PH * 2);
  __bf16* s1   = (__bf16*)alloc((size_t)PM * PH * 2);
  float*  sc   = (float*) alloc((size_t)PM * PH * 4);
  __bf16* Y    = (__bf16*)alloc((size_t)PM * PH * 2);
  unsigned* hist1 = (unsigned*)alloc(PB * 4096 * 4);
  unsigned* hist2 = (unsigned*)alloc(PB * 4096 * 4);
  unsigned* hist3 = (unsigned*)alloc(PB * 256 * 4);
  unsigned* state = (unsigned*)alloc(PB * 2 * 4);
  int*      mask  = (int*)     alloc(PB * PH * 4);

  struct Conv { const float* s; __bf16* d; size_t n; };
  const Conv convs[6] = {
    { inputs,  Xbf,  (size_t)PM * PD },
    { gate_w1, g1bf, (size_t)PH * PD },
    { gate_w2, g2bf, (size_t)PH * PH },
    { up_w,    upbf, (size_t)PH * PD },
    { mod_w,   mdbf, (size_t)PH * PD },
    { down_w,  dnbf, (size_t)PD * PH },
  };
  for (int c = 0; c < 6; ++c) {
    unsigned blocks = (unsigned)((convs[c].n + 255) / 256);
    f2bf_kernel<<<blocks, 256, 0, stream>>>(convs[c].s, convs[c].d, convs[c].n);
  }

  // GEMM1: s1 = relu(X @ G1^T + b1)   [8192,4096], K=1024
  gemm128_bf16<MODE_RELU_BF16><<<dim3(PH / 128, PM / 128), 256, 0, stream>>>(
      Xbf, g1bf, nullptr, gate_b1, nullptr, nullptr, (void*)s1, PM, PH, PD);

  // GEMM2: scores = s1 @ G2^T + b2    [8192,4096], K=4096
  gemm128_bf16<MODE_SCORES_F32><<<dim3(PH / 128, PM / 128), 256, 0, stream>>>(
      s1, g2bf, nullptr, gate_b2, nullptr, nullptr, (void*)sc, PM, PH, PH);

  // Radix select of per-batch k-th largest score (12+12+8 key bits)
  hipMemsetAsync(hist1, 0, PB * 4096 * 4, stream);
  hipMemsetAsync(hist2, 0, PB * 4096 * 4, stream);
  hipMemsetAsync(hist3, 0, PB * 256 * 4, stream);
  init_state<<<1, 32, 0, stream>>>(state);

  hist_top12<<<dim3(PB, 32), 1024, 0, stream>>>(sc, hist1);
  scan_hist<<<PB, 32, 0, stream>>>(hist1, state, 4096, 12);
  refine_hist<<<dim3(PB, 32), 1024, 0, stream>>>(sc, state, hist2, 20, 8, 0xFFFu);
  scan_hist<<<PB, 32, 0, stream>>>(hist2, state, 4096, 12);
  refine_hist<<<dim3(PB, 32), 1024, 0, stream>>>(sc, state, hist3, 8, 0, 0xFFu);
  scan_hist<<<PB, 32, 0, stream>>>(hist3, state, 256, 8);

  colmax_mask<<<dim3(PH / 256, PB), 256, 0, stream>>>(sc, state, mask);

  // GEMM3: Y = silu(X @ Wsel^T + bsel)
  gemm128_bf16<MODE_SELECT_SILU><<<dim3(PH / 128, PM / 128), 256, 0, stream>>>(
      Xbf, upbf, mdbf, up_b, mod_b, mask, (void*)Y, PM, PH, PD);

  // GEMM4: out = Y @ down_w^T + down_b   [8192,1024], K=4096
  gemm128_bf16<MODE_PLAIN_F32><<<dim3(PD / 128, PM / 128), 256, 0, stream>>>(
      Y, dnbf, nullptr, down_b, nullptr, nullptr, (void*)out, PM, PD, PH);
}
